// EGADLayer_67156108640607
// MI455X (gfx1250) — matrix operand order repaired and checked by reading, not compiled, not run on hardware
//
#include <hip/hip_runtime.h>
#include <hip/hip_bf16.h>
#include <math.h>

#define B_ROWS 4096
#define U_COLS 8192
#define DDIM   256
#define CHUNK  128

typedef __bf16 bf16_t;
typedef __attribute__((ext_vector_type(16))) __bf16        bf16x16;
typedef __attribute__((ext_vector_type(8)))  float         f32x8;
typedef __attribute__((ext_vector_type(4)))  unsigned int  u32x4;
typedef __attribute__((ext_vector_type(4)))  int           i32x4;
typedef __attribute__((address_space(1))) i32x4* as1_i32x4_p;
typedef __attribute__((address_space(3))) i32x4* as3_i32x4_p;

#if __has_builtin(__builtin_amdgcn_global_load_async_to_lds_b128) && \
    __has_builtin(__builtin_amdgcn_s_wait_asynccnt)
#define USE_ASYNC_LDS 1
#else
#define USE_ASYNC_LDS 0
#endif

union FragU { u32x4 q[2]; bf16x16 v; };

__device__ __forceinline__ f32x8 wmma_bf16(bf16x16 a, bf16x16 b, f32x8 c) {
  return __builtin_amdgcn_wmma_f32_16x16x32_bf16(false, a, false, b, (short)0, c,
                                                 false, false);
}

// A-matrix fragment: 16x32 bf16 tile, row-major source (stride in elements).
// Lanes 0-15 rows 0-15 with K = kb..kb+15 (kb=0); lanes 16-31 kb=16.
__device__ __forceinline__ bf16x16 load_fragA(const bf16_t* base, int stride,
                                              int kOff, int lane) {
  int r  = lane & 15;
  int kb = kOff + ((lane >> 4) << 4);
  const bf16_t* p = base + (size_t)r * stride + kb;
  FragU f;
  f.q[0] = *reinterpret_cast<const u32x4*>(p);
  f.q[1] = *reinterpret_cast<const u32x4*>(p + 8);
  return f.v;
}

// B-matrix fragment: 32x16 bf16 tile where B[k][n] = src[n*stride + k].
// Lane n holds K = kb..kb+15, kb = (lane>=16) ? 16 : 0.
__device__ __forceinline__ bf16x16 load_fragB(const bf16_t* base, int stride,
                                              int kOff, int lane) {
  int n  = lane & 15;
  int kb = kOff + ((lane >> 4) << 4);
  const bf16_t* p = base + (size_t)n * stride + kb;
  FragU f;
  f.q[0] = *reinterpret_cast<const u32x4*>(p);
  f.q[1] = *reinterpret_cast<const u32x4*>(p + 8);
  return f.v;
}

// B-matrix fragment where source is K-major rows: B[k][n] = src[k*stride + dCol + n].
__device__ __forceinline__ bf16x16 load_fragV(const bf16_t* src, int stride,
                                              int kOff, int dCol, int lane) {
  int n  = lane & 15;
  int kb = kOff + ((lane >> 4) << 4);
  bf16x16 v;
  #pragma unroll
  for (int j = 0; j < 16; ++j)
    v[j] = src[(size_t)(kb + j) * stride + dCol + n];
  return v;
}

// ---- weight convert + transpose: WT[n][k] = (bf16)W[k][n] --------------------
__global__ __launch_bounds__(256) void convw_kernel(
    const float* __restrict__ We, const float* __restrict__ Wv,
    const float* __restrict__ fcW, bf16_t* __restrict__ WeT,
    bf16_t* __restrict__ WvT, bf16_t* __restrict__ fcWT) {
  int tid = blockIdx.x * 256 + threadIdx.x;   // 3 * 256 * 256 threads total
  int which = tid >> 16;
  int rem = tid & 65535;
  int k = rem >> 8, n = rem & 255;
  const float* W = (which == 0) ? We : (which == 1) ? Wv : fcW;
  bf16_t* WT = (which == 0) ? WeT : (which == 1) ? WvT : fcWT;
  WT[(size_t)n * DDIM + k] = (bf16_t)W[(size_t)k * DDIM + n];
}

// ---- row gathers + per-row norms / sumsq ------------------------------------
__global__ __launch_bounds__(256) void gather_kernel(
    const float* __restrict__ node_embed, const float* __restrict__ edge_embed,
    const int* __restrict__ nodes, const int* __restrict__ uedges,
    const int* __restrict__ unodes, float* __restrict__ nodematF,
    bf16_t* __restrict__ nodematB, float* __restrict__ nodeSq,
    bf16_t* __restrict__ eeB, float* __restrict__ enorm,
    bf16_t* __restrict__ nmatB, float* __restrict__ nmatSq) {
  __shared__ float red[256];
  int row = blockIdx.x, t = threadIdx.x;
  float v;
  if (row < B_ROWS) {
    int idx = nodes[row];
    v = node_embed[(size_t)idx * DDIM + t];
    nodematF[(size_t)row * DDIM + t] = v;
    nodematB[(size_t)row * DDIM + t] = (bf16_t)v;
  } else if (row < B_ROWS + U_COLS) {
    int r = row - B_ROWS;
    int idx = uedges[r];
    v = edge_embed[(size_t)idx * DDIM + t];
    eeB[(size_t)r * DDIM + t] = (bf16_t)v;
  } else {
    int r = row - B_ROWS - U_COLS;
    int idx = unodes[r];
    v = node_embed[(size_t)idx * DDIM + t];
    nmatB[(size_t)r * DDIM + t] = (bf16_t)v;
  }
  red[t] = v * v;
  __syncthreads();
  #pragma unroll
  for (int s = 128; s > 0; s >>= 1) {
    if (t < s) red[t] += red[t + s];
    __syncthreads();
  }
  if (t == 0) {
    float ss = red[0];
    if (row < B_ROWS) nodeSq[row] = ss;
    else if (row < B_ROWS + U_COLS) enorm[row - B_ROWS] = sqrtf(ss);
    else nmatSq[row - B_ROWS - U_COLS] = ss;
  }
}

// ---- (M,256) @ (256,256) WMMA GEMM with fused epilogue ----------------------
// flags: bit0 = leaky relu, bit1 = nan -> 0.01 on the raw GEMM result
__global__ __launch_bounds__(256) void gemm256_kernel(
    const bf16_t* __restrict__ A, const bf16_t* __restrict__ WT,
    const float* __restrict__ bias, const float* __restrict__ add1,
    const float* __restrict__ add2, float* __restrict__ outF,
    bf16_t* __restrict__ outB, float* __restrict__ rowNorm, int flags) {
  __shared__ float part[8][16];
  int tid = threadIdx.x;
  int lane = tid & 31;
  int w = tid >> 5;
  int m0 = blockIdx.x << 4;
  int n0 = w << 5;

  bf16x16 afr[8];
  #pragma unroll
  for (int k = 0; k < 8; ++k)
    afr[k] = load_fragA(A + (size_t)m0 * DDIM, DDIM, k * 32, lane);

  f32x8 acc0 = {}, acc1 = {};
  #pragma unroll
  for (int k = 0; k < 8; ++k) {
    bf16x16 b0 = load_fragB(WT + (size_t)n0 * DDIM, DDIM, k * 32, lane);
    acc0 = wmma_bf16(afr[k], b0, acc0);
    bf16x16 b1 = load_fragB(WT + (size_t)(n0 + 16) * DDIM, DDIM, k * 32, lane);
    acc1 = wmma_bf16(afr[k], b1, acc1);
  }

  int n = lane & 15;
  int hb = (lane >> 4) << 3;
  #pragma unroll
  for (int r = 0; r < 8; ++r) {
    int m = r + hb;
    size_t rowOff = (size_t)(m0 + m) * DDIM;
    float v0 = acc0[r], v1 = acc1[r];
    if (flags & 2) {
      if (v0 != v0) v0 = 0.01f;
      if (v1 != v1) v1 = 0.01f;
    }
    int c0 = n0 + n, c1 = n0 + 16 + n;
    if (add1) { v0 += add1[rowOff + c0]; v1 += add1[rowOff + c1]; }
    if (add2) { v0 += add2[rowOff + c0]; v1 += add2[rowOff + c1]; }
    if (bias) { v0 += bias[c0]; v1 += bias[c1]; }
    if (flags & 1) {
      v0 = (v0 > 0.f) ? v0 : 0.01f * v0;
      v1 = (v1 > 0.f) ? v1 : 0.01f * v1;
    }
    if (outF) { outF[rowOff + c0] = v0; outF[rowOff + c1] = v1; }
    if (outB) { outB[rowOff + c0] = (bf16_t)v0; outB[rowOff + c1] = (bf16_t)v1; }
    if (rowNorm) {
      float s = v0 * v0 + v1 * v1;            // deterministic 16-lane reduce
      s += __shfl_xor(s, 1);
      s += __shfl_xor(s, 2);
      s += __shfl_xor(s, 4);
      s += __shfl_xor(s, 8);
      if (n == 0) part[w][m] = s;
    }
  }
  if (rowNorm) {
    __syncthreads();
    if (tid < 16) {
      float s = 0.f;
      #pragma unroll
      for (int ww = 0; ww < 8; ++ww) s += part[ww][tid];
      rowNorm[m0 + tid] = sqrtf(s);
    }
  }
}

// ---- fused masked attention with online softmax -----------------------------
// mode 0 (edge):  score = (1 - dot/(rowv[m]*colv[u])) * mask
// mode 1 (node):  score = sqrt(max(rowv[m] + colv[u] - 2*dot, 1e-12)) * mask
__global__ __launch_bounds__(256) void attn_kernel(
    const bf16_t* __restrict__ Abf, const bf16_t* __restrict__ Ebf,
    const float* __restrict__ rowvG, const float* __restrict__ colvG,
    const int* __restrict__ mask, float* __restrict__ outF,
    bf16_t* __restrict__ outB, int mode) {
  __shared__ __align__(16) bf16_t Ebuf[CHUNK * DDIM];   // 64 KB
  __shared__ float Sbuf[16 * CHUNK];                    // 8 KB
  __shared__ __align__(16) bf16_t Pbuf[16 * CHUNK];     // 4 KB
  __shared__ float rowM[16], rowL[16], rowScale[16], rowVal[16];
  __shared__ float colv[CHUNK];

  int tid = threadIdx.x;
  int lane = tid & 31;
  int w = tid >> 5;
  int m0 = blockIdx.x << 4;
  int d0 = w << 5;
  int n = lane & 15;
  int hb = (lane >> 4) << 3;

  if (tid < 16) {
    rowM[tid] = -3.0e38f;
    rowL[tid] = 0.f;
    rowVal[tid] = rowvG[m0 + tid];
  }

  bf16x16 afr[8];
  #pragma unroll
  for (int k = 0; k < 8; ++k)
    afr[k] = load_fragA(Abf + (size_t)m0 * DDIM, DDIM, k * 32, lane);

  f32x8 acc0 = {}, acc1 = {};

  for (int c0 = 0; c0 < U_COLS; c0 += CHUNK) {
    __syncthreads();                       // Ebuf/Sbuf reuse guard
#if USE_ASYNC_LDS
    {                                      // async DMA: global -> LDS, b128
      const char* srcB = (const char*)(Ebf + (size_t)c0 * DDIM);
      unsigned ldsBase = (unsigned)(size_t)(const void*)Ebuf;
      #pragma unroll
      for (int j = 0; j < (CHUNK * DDIM * 2 / 16) / 256; ++j) {
        unsigned idx = (unsigned)(tid + j * 256) * 16u;
        __builtin_amdgcn_global_load_async_to_lds_b128(
            (as1_i32x4_p)(unsigned long long)(size_t)(srcB + idx),
            (as3_i32x4_p)(unsigned)(ldsBase + idx), 0, 0);
      }
      if (tid < CHUNK) colv[tid] = colvG[c0 + tid];
      __builtin_amdgcn_s_wait_asynccnt(0);
    }
#else
    {                                      // manual stage via VGPRs
      const u32x4* src = reinterpret_cast<const u32x4*>(Ebf + (size_t)c0 * DDIM);
      u32x4* dst = reinterpret_cast<u32x4*>(Ebuf);
      #pragma unroll
      for (int j = 0; j < (CHUNK * DDIM / 8) / 256; ++j)
        dst[tid + j * 256] = src[tid + j * 256];
      if (tid < CHUNK) colv[tid] = colvG[c0 + tid];
    }
#endif
    __syncthreads();

    {                                      // score tile: 16 x 16 per wave
      f32x8 cs = {};
      #pragma unroll
      for (int k = 0; k < 8; ++k) {
        bf16x16 bfr = load_fragB(Ebuf + (size_t)(w * 16) * DDIM, DDIM, k * 32, lane);
        cs = wmma_bf16(afr[k], bfr, cs);
      }
      int uloc = w * 16 + n;
      const int* mcol = mask + (size_t)m0 * U_COLS + c0 + uloc;
      if (c0 + CHUNK < U_COLS) __builtin_prefetch(mcol + CHUNK, 0, 1);
      #pragma unroll
      for (int r = 0; r < 8; ++r) {
        int m = r + hb;
        float dot = cs[r];
        float s;
        if (mode == 0) {
          s = 1.0f - dot / (rowVal[m] * colv[uloc]);
        } else {
          float d2 = rowVal[m] + colv[uloc] - 2.0f * dot;
          s = sqrtf(fmaxf(d2, 1e-12f));
        }
        s *= (float)mcol[(size_t)m * U_COLS];
        Sbuf[m * CHUNK + uloc] = s;
      }
    }
    __syncthreads();

    if (w == 0) {                          // online softmax stats, one wave
      int row = lane & 15;
      const float* srow = Sbuf + row * CHUNK + ((lane >> 4) << 6);
      float mOld = rowM[row];
      float mx = -3.0e38f;
      #pragma unroll 8
      for (int j = 0; j < 64; ++j) mx = fmaxf(mx, srow[j]);
      mx = fmaxf(mx, __shfl_xor(mx, 16));
      float mNew = fmaxf(mOld, mx);
      float sum = 0.f;
      #pragma unroll 8
      for (int j = 0; j < 64; ++j) sum += __expf(srow[j] - mNew);
      sum += __shfl_xor(sum, 16);
      if (lane < 16) {
        float f = __expf(mOld - mNew);
        rowScale[row] = f;
        rowL[row] = rowL[row] * f + sum;
        rowM[row] = mNew;
      }
    }
    __syncthreads();

    #pragma unroll                         // probabilities -> bf16
    for (int j = 0; j < (16 * CHUNK) / 256; ++j) {
      int idx = tid + j * 256;
      int m = idx >> 7;                    // / CHUNK
      Pbuf[idx] = (bf16_t)__expf(Sbuf[idx] - rowM[m]);
    }
    #pragma unroll                         // rescale running accumulators
    for (int r = 0; r < 8; ++r) {
      float f = rowScale[r + hb];
      acc0[r] *= f;
      acc1[r] *= f;
    }
    __syncthreads();

    #pragma unroll                         // acc += P(16xCHUNK) x V(CHUNKx32)
    for (int k = 0; k < CHUNK / 32; ++k) {
      bf16x16 pfr = load_fragA(Pbuf, CHUNK, k * 32, lane);
      bf16x16 v0 = load_fragV(Ebuf, DDIM, k * 32, d0, lane);
      acc0 = wmma_bf16(pfr, v0, acc0);
      bf16x16 v1 = load_fragV(Ebuf, DDIM, k * 32, d0 + 16, lane);
      acc1 = wmma_bf16(pfr, v1, acc1);
    }
  }

  #pragma unroll
  for (int r = 0; r < 8; ++r) {
    int m = r + hb;
    float inv = 1.0f / rowL[m];
    float v0 = acc0[r] * inv;
    float v1 = acc1[r] * inv;
    if (v0 != v0) v0 = 0.01f;
    if (v1 != v1) v1 = 0.01f;
    size_t rowOff = (size_t)(m0 + m) * DDIM;
    if (outF) { outF[rowOff + d0 + n] = v0; outF[rowOff + d0 + 16 + n] = v1; }
    if (outB) {
      outB[rowOff + d0 + n] = (bf16_t)v0;
      outB[rowOff + d0 + 16 + n] = (bf16_t)v1;
    }
  }
}

extern "C" void kernel_launch(void* const* d_in, const int* in_sizes, int n_in,
                              void* d_out, int out_size, void* d_ws,
                              size_t ws_size, hipStream_t stream) {
  const float* node_embed = (const float*)d_in[0];
  const float* edge_embed = (const float*)d_in[1];
  const float* W_e  = (const float*)d_in[2];
  const float* W_v  = (const float*)d_in[3];
  const float* fc_W = (const float*)d_in[4];
  const float* fc_b = (const float*)d_in[5];
  const int* nodes  = (const int*)d_in[6];
  const int* uedges = (const int*)d_in[7];
  const int* unodes = (const int*)d_in[8];
  const int* mask_e = (const int*)d_in[9];
  const int* mask_n = (const int*)d_in[10];
  float* out = (float*)d_out;
  (void)in_sizes; (void)n_in; (void)out_size; (void)ws_size;

  char* ws = (char*)d_ws;
  size_t off = 0;
  auto alloc = [&](size_t bytes) -> char* {
    char* p = ws + off;
    off = (off + bytes + 255) & ~(size_t)255;
    return p;
  };
  float*  nodematF = (float*) alloc((size_t)B_ROWS * DDIM * 4);
  bf16_t* nodematB = (bf16_t*)alloc((size_t)B_ROWS * DDIM * 2);
  float*  nodeSq   = (float*) alloc((size_t)B_ROWS * 4);
  bf16_t* nmeB     = (bf16_t*)alloc((size_t)B_ROWS * DDIM * 2);
  float*  nnorm    = (float*) alloc((size_t)B_ROWS * 4);
  bf16_t* eeB      = (bf16_t*)alloc((size_t)U_COLS * DDIM * 2);
  float*  enorm    = (float*) alloc((size_t)U_COLS * 4);
  bf16_t* nmatB    = (bf16_t*)alloc((size_t)U_COLS * DDIM * 2);
  float*  nmatSq   = (float*) alloc((size_t)U_COLS * 4);
  bf16_t* WeT      = (bf16_t*)alloc((size_t)DDIM * DDIM * 2);
  bf16_t* WvT      = (bf16_t*)alloc((size_t)DDIM * DDIM * 2);
  bf16_t* fcWT     = (bf16_t*)alloc((size_t)DDIM * DDIM * 2);
  bf16_t* O1B      = (bf16_t*)alloc((size_t)B_ROWS * DDIM * 2);
  float*  O2F      = (float*) alloc((size_t)B_ROWS * DDIM * 4);
  bf16_t* embB     = (bf16_t*)alloc((size_t)B_ROWS * DDIM * 2);

  convw_kernel<<<(3 * DDIM * DDIM) / 256, 256, 0, stream>>>(W_e, W_v, fc_W, WeT,
                                                            WvT, fcWT);
  gather_kernel<<<B_ROWS + 2 * U_COLS, 256, 0, stream>>>(
      node_embed, edge_embed, nodes, uedges, unodes, nodematF, nodematB, nodeSq,
      eeB, enorm, nmatB, nmatSq);
  // nm_e = node_matrix @ W_e   (bf16 copy + row norms)
  gemm256_kernel<<<B_ROWS / 16, 256, 0, stream>>>(
      nodematB, WeT, nullptr, nullptr, nullptr, nullptr, nmeB, nnorm, 0);
  // edge attention -> O1 = attn_e @ ee   (bf16)
  attn_kernel<<<B_ROWS / 16, 256, 0, stream>>>(nmeB, eeB, nnorm, enorm, mask_e,
                                               nullptr, O1B, 0);
  // node attention -> O2 = nn_feat      (f32)
  attn_kernel<<<B_ROWS / 16, 256, 0, stream>>>(nodematB, nmatB, nodeSq, nmatSq,
                                               mask_n, O2F, nullptr, 1);
  // emb = nan2num(O1 @ W_v) + node_matrix + O2   (bf16)
  gemm256_kernel<<<B_ROWS / 16, 256, 0, stream>>>(
      O1B, WvT, nullptr, nodematF, O2F, nullptr, embB, nullptr, 2);
  // out = leaky_relu(emb @ fc_W + fc_b)
  gemm256_kernel<<<B_ROWS / 16, 256, 0, stream>>>(
      embB, fcWT, fc_b, nullptr, nullptr, out, nullptr, nullptr, 1);
}
